// Attention_19997367730710
// MI455X (gfx1250) — matrix operand order repaired and checked by reading, not compiled, not run on hardware
//
#include <hip/hip_runtime.h>

#define DIM    512
#define HEADS  8
#define DHEAD  64
#define INNER  512
#define NQKV   1536
#define BATCH  4
#define SEQ    2048
#define M_TOT  (BATCH * SEQ)   // 8192
#define EPSV   1e-5f

typedef __bf16 bf16;
typedef __attribute__((ext_vector_type(16))) __bf16 v16bf;
typedef __attribute__((ext_vector_type(8)))  __bf16 v8bf;
typedef __attribute__((ext_vector_type(8)))  float  v8f;

#if __has_builtin(__builtin_amdgcn_global_load_async_to_lds_b128) && \
    __has_builtin(__builtin_amdgcn_s_wait_asynccnt)
#define USE_ASYNC_LDS 1
typedef int v4i __attribute__((vector_size(16)));
typedef __attribute__((address_space(1))) v4i as1_v4i;
typedef __attribute__((address_space(3))) v4i as3_v4i;
#else
#define USE_ASYNC_LDS 0
#endif

// Build a 16-element bf16 A/B fragment from two contiguous 16-byte chunks.
// Per CDNA5 ISA: 16-bit A 16x32 -> lane (m + 16s): elems 0..7 = A[m, 8s+0..7],
// elems 8..15 = A[m, 16+8s+0..7]. B 32x16 -> lane (n + 16s): same k order as A, B[k, n]
// (two 16B runs per lane -> lo = base+8s, hi = base+16+8s).
__device__ __forceinline__ v16bf load_frag(const bf16* lo, const bf16* hi) {
  v16bf r;
  v8bf a = *(const v8bf*)lo;
  v8bf b = *(const v8bf*)hi;
#pragma unroll
  for (int i = 0; i < 8; ++i) { r[i] = a[i]; r[i + 8] = b[i]; }
  return r;
}

// Reductions across the 16 lanes sharing the same s-half (xor masks < 16 stay
// within the half in wave32).
__device__ __forceinline__ float grp16_max(float v) {
#pragma unroll
  for (int off = 1; off < 16; off <<= 1) v = fmaxf(v, __shfl_xor(v, off, 32));
  return v;
}
__device__ __forceinline__ float grp16_sum(float v) {
#pragma unroll
  for (int off = 1; off < 16; off <<= 1) v += __shfl_xor(v, off, 32);
  return v;
}

// ---------------------------------------------------------------- LayerNorm(dim)
__global__ __launch_bounds__(256) void k_ln(const float* __restrict__ x,
                                            const float* __restrict__ g,
                                            const float* __restrict__ b,
                                            bf16* __restrict__ xn) {
  int wave = threadIdx.x >> 5, lane = threadIdx.x & 31;
  int row = blockIdx.x * 8 + wave;
  const float* xr = x + (size_t)row * DIM + lane * 16;
  float v[16]; float s = 0.f, s2 = 0.f;
#pragma unroll
  for (int i = 0; i < 16; i += 4) {
    float4 t = *(const float4*)(xr + i);
    v[i] = t.x; v[i+1] = t.y; v[i+2] = t.z; v[i+3] = t.w;
  }
#pragma unroll
  for (int i = 0; i < 16; ++i) { s += v[i]; s2 += v[i] * v[i]; }
#pragma unroll
  for (int off = 1; off < 32; off <<= 1) {
    s  += __shfl_xor(s,  off, 32);
    s2 += __shfl_xor(s2, off, 32);
  }
  float mu   = s * (1.f / DIM);
  float var  = s2 * (1.f / DIM) - mu * mu;
  float rstd = rsqrtf(var + EPSV);
  bf16* o = xn + (size_t)row * DIM + lane * 16;
#pragma unroll
  for (int i = 0; i < 16; ++i) {
    int c = lane * 16 + i;
    o[i] = (bf16)((v[i] - mu) * rstd * g[c] + b[c]);
  }
}

// ---------------------------------------- transpose f32 weights -> bf16 [o][c]
__global__ void k_transpose_bf16(const float* __restrict__ src, bf16* __restrict__ dst,
                                 int rows /*c*/, int cols /*o*/) {
  int total = rows * cols;
  for (int idx = blockIdx.x * blockDim.x + threadIdx.x; idx < total;
       idx += gridDim.x * blockDim.x) {
    int o = idx / rows, c = idx - o * rows;
    dst[idx] = (bf16)src[(size_t)c * cols + o];   // dst[o*rows + c] = src[c][o]
  }
}

// ------------------------------------------- QKV GEMM (WMMA, wave tile 32x64)
// q,k stored [h][M][d] row-major; v stored transposed [h][d][M].
// When available, stages A/B tiles through LDS with double-buffered
// GLOBAL_LOAD_ASYNC_TO_LDS_B128 (ASYNCcnt pipelining); otherwise direct loads.
__global__ __launch_bounds__(32) void k_qkv(const bf16* __restrict__ xn,
                                            const bf16* __restrict__ wT,
                                            bf16* __restrict__ q, bf16* __restrict__ k,
                                            bf16* __restrict__ vT) {
  int lane = threadIdx.x & 31;
  int np = lane & 15, s = lane >> 4;
  int m0 = blockIdx.x * 32;
  int o0 = blockIdx.y * 64;
  v8f acc[2][4] = {};

#if USE_ASYNC_LDS
  __shared__ alignas(16) bf16 sa[2][32][32];    // A tile: 32 rows x 32 k
  __shared__ alignas(16) bf16 sw[2][64][32];    // B tile: 64 cols x 32 k
  auto stage = [&](int buf, int kc) {
#pragma unroll
    for (int t = 0; t < 4; ++t) {           // A: 2KB = 128 x 16B chunks
      int c = lane + 32 * t;
      int row = c >> 2, ko = (c & 3) * 8;
      __builtin_amdgcn_global_load_async_to_lds_b128(
          (as1_v4i*)(size_t)(xn + (size_t)(m0 + row) * DIM + kc + ko),
          (as3_v4i*)&sa[buf][row][ko], 0, 0);
    }
#pragma unroll
    for (int t = 0; t < 8; ++t) {           // B: 4KB = 256 x 16B chunks
      int c = lane + 32 * t;
      int col = c >> 2, ko = (c & 3) * 8;
      __builtin_amdgcn_global_load_async_to_lds_b128(
          (as1_v4i*)(size_t)(wT + (size_t)(o0 + col) * DIM + kc + ko),
          (as3_v4i*)&sw[buf][col][ko], 0, 0);
    }
  };
  stage(0, 0);
#pragma unroll
  for (int ki = 0; ki < DIM / 32; ++ki) {
    int buf = ki & 1;
    if (ki + 1 < DIM / 32) {
      stage(buf ^ 1, (ki + 1) * 32);
      __builtin_amdgcn_s_wait_asynccnt(12);  // in-order: current tile resident
    } else {
      __builtin_amdgcn_s_wait_asynccnt(0);
    }
    v16bf a[2], bf[4];
#pragma unroll
    for (int ms = 0; ms < 2; ++ms) {
      const bf16* base = &sa[buf][ms * 16 + np][0];
      a[ms] = load_frag(base + s * 8, base + 16 + s * 8);
    }
#pragma unroll
    for (int ns = 0; ns < 4; ++ns) {
      const bf16* base = &sw[buf][ns * 16 + np][0];
      bf[ns] = load_frag(base + s * 8, base + 16 + s * 8);
    }
#pragma unroll
    for (int ms = 0; ms < 2; ++ms)
#pragma unroll
      for (int ns = 0; ns < 4; ++ns)
        acc[ms][ns] = __builtin_amdgcn_wmma_f32_16x16x32_bf16(
            false, a[ms], false, bf[ns], (short)0, acc[ms][ns], false, false);
  }
#else
  for (int kc = 0; kc < DIM; kc += 32) {
    v16bf a[2], bf[4];
#pragma unroll
    for (int ms = 0; ms < 2; ++ms) {
      const bf16* base = xn + (size_t)(m0 + ms * 16 + np) * DIM + kc;
      a[ms] = load_frag(base + s * 8, base + 16 + s * 8);
    }
#pragma unroll
    for (int ns = 0; ns < 4; ++ns) {
      const bf16* base = wT + (size_t)(o0 + ns * 16 + np) * DIM + kc;
      bf[ns] = load_frag(base + s * 8, base + 16 + s * 8);
    }
#pragma unroll
    for (int ms = 0; ms < 2; ++ms)
#pragma unroll
      for (int ns = 0; ns < 4; ++ns)
        acc[ms][ns] = __builtin_amdgcn_wmma_f32_16x16x32_bf16(
            false, a[ms], false, bf[ns], (short)0, acc[ms][ns], false, false);
  }
#endif

  // Store. sec/hd/dcbase are wave-uniform per ns (16-col subtiles never straddle
  // the 512-wide q/k/v sections nor a 64-wide head column block).
#pragma unroll
  for (int ns = 0; ns < 4; ++ns) {
    int colbase = o0 + ns * 16;          // uniform
    int sec = colbase >> 9;              // 0=q 1=k 2=v (uniform)
    int hd  = (colbase & 511) >> 6;      // uniform
    int dc  = (colbase & 63) + np;       // per-lane column within head
    if (sec == 2) {
      // v: transposed layout -> 8 consecutive rows per lane = one 16B store
#pragma unroll
      for (int ms = 0; ms < 2; ++ms) {
        int row0 = m0 + ms * 16 + s * 8;
        v8bf pk;
#pragma unroll
        for (int r = 0; r < 8; ++r) pk[r] = (bf16)acc[ms][ns][r];
        *(v8bf*)(vT + ((size_t)hd * DHEAD + dc) * M_TOT + row0) = pk;
      }
    } else {
      bf16* dst = (sec == 0) ? q : k;
#pragma unroll
      for (int ms = 0; ms < 2; ++ms)
#pragma unroll
        for (int r = 0; r < 8; ++r) {
          int row = m0 + ms * 16 + s * 8 + r;
          dst[((size_t)hd * M_TOT + row) * DHEAD + dc] = (bf16)acc[ms][ns][r];
        }
    }
  }
}

// ------------------------- fused two-pass attention + re-attn mix + head-LN + @V
// Block = 8 waves; wave h owns head h; 32 query rows per block.
__global__ __launch_bounds__(256) void k_attn(const bf16* __restrict__ q,
                                              const bf16* __restrict__ k,
                                              const bf16* __restrict__ vT,
                                              const float* __restrict__ reattn_w,
                                              const float* __restrict__ rn_g,
                                              const float* __restrict__ rn_b,
                                              bf16* __restrict__ attn_out) {
  __shared__ float lds_p[HEADS][32][33];
  __shared__ float Wmix[HEADS][HEADS];
  __shared__ float sg[HEADS], sb[HEADS];
  int tid = threadIdx.x;
  if (tid < 64) Wmix[tid >> 3][tid & 7] = reattn_w[tid];
  if (tid < 8) { sg[tid] = rn_g[tid]; sb[tid] = rn_b[tid]; }
  __syncthreads();

  int lane = tid & 31, h = tid >> 5;
  int np = lane & 15, s = lane >> 4;
  int m0 = blockIdx.x * 32;
  int jbase = (m0 / SEQ) * SEQ;                 // stay inside this batch
  const bf16* Qh = q  + (size_t)h * M_TOT * DHEAD;
  const bf16* Kh = k  + (size_t)h * M_TOT * DHEAD;
  const bf16* Vh = vT + (size_t)h * DHEAD * M_TOT;
  const float scale = 0.125f;                   // 64^-0.5

  v16bf aq[2][2];
#pragma unroll
  for (int ms = 0; ms < 2; ++ms)
#pragma unroll
    for (int kd = 0; kd < 2; ++kd) {
      const bf16* base = Qh + (size_t)(m0 + ms * 16 + np) * DHEAD + kd * 32;
      aq[ms][kd] = load_frag(base + s * 8, base + 16 + s * 8);
    }

  float rmax[2][8], rsum[2][8];
#pragma unroll
  for (int ms = 0; ms < 2; ++ms)
#pragma unroll
    for (int r = 0; r < 8; ++r) { rmax[ms][r] = -3.0e38f; rsum[ms][r] = 0.f; }

  // ---- pass 1: online softmax stats over full key range
  for (int jt = 0; jt < SEQ / 16; ++jt) {
    int j0 = jbase + jt * 16;
    __builtin_prefetch(Kh + (size_t)(j0 + 16 + np) * DHEAD, 0, 1);
    v16bf bk[2];
#pragma unroll
    for (int kd = 0; kd < 2; ++kd) {
      const bf16* base = Kh + (size_t)(j0 + np) * DHEAD + kd * 32;
      bk[kd] = load_frag(base + s * 8, base + 16 + s * 8);
    }
#pragma unroll
    for (int ms = 0; ms < 2; ++ms) {
      v8f d = {};
      d = __builtin_amdgcn_wmma_f32_16x16x32_bf16(false, aq[ms][0], false, bk[0], (short)0, d, false, false);
      d = __builtin_amdgcn_wmma_f32_16x16x32_bf16(false, aq[ms][1], false, bk[1], (short)0, d, false, false);
#pragma unroll
      for (int r = 0; r < 8; ++r) {
        float val  = d[r] * scale;
        float tmax = grp16_max(val);
        float nm   = fmaxf(rmax[ms][r], tmax);
        float tsum = grp16_sum(__expf(val - nm));
        rsum[ms][r] = rsum[ms][r] * __expf(rmax[ms][r] - nm) + tsum;
        rmax[ms][r] = nm;
      }
    }
  }
  float rinv[2][8];
#pragma unroll
  for (int ms = 0; ms < 2; ++ms)
#pragma unroll
    for (int r = 0; r < 8; ++r) rinv[ms][r] = 1.f / rsum[ms][r];

  // ---- pass 2: recompute dots, normalize, mix heads, head-LN, @ V
  v8f acc[2][4] = {};
  for (int jt = 0; jt < SEQ / 32; ++jt) {
    int j0 = jbase + jt * 32;
#pragma unroll
    for (int js = 0; js < 2; ++js) {
      v16bf bk[2];
#pragma unroll
      for (int kd = 0; kd < 2; ++kd) {
        const bf16* base = Kh + (size_t)(j0 + js * 16 + np) * DHEAD + kd * 32;
        bk[kd] = load_frag(base + s * 8, base + 16 + s * 8);
      }
#pragma unroll
      for (int ms = 0; ms < 2; ++ms) {
        v8f d = {};
        d = __builtin_amdgcn_wmma_f32_16x16x32_bf16(false, aq[ms][0], false, bk[0], (short)0, d, false, false);
        d = __builtin_amdgcn_wmma_f32_16x16x32_bf16(false, aq[ms][1], false, bk[1], (short)0, d, false, false);
#pragma unroll
        for (int r = 0; r < 8; ++r) {
          float val = d[r] * scale;
          lds_p[h][ms * 16 + s * 8 + r][js * 16 + np] =
              __expf(val - rmax[ms][r]) * rinv[ms][r];
        }
      }
    }
    __syncthreads();
    // Rebuild this head's A' fragment (A layout) from all 8 heads' probs.
    v16bf ap[2];
#pragma unroll
    for (int ms = 0; ms < 2; ++ms) {
      int irow = ms * 16 + np;
#pragma unroll
      for (int e = 0; e < 16; ++e) {
        int jc = (e < 8) ? (s * 8 + e) : (16 + s * 8 + (e - 8));
        float mix[HEADS];
#pragma unroll
        for (int gg = 0; gg < HEADS; ++gg) mix[gg] = 0.f;
#pragma unroll
        for (int hh = 0; hh < HEADS; ++hh) {
          float pv = lds_p[hh][irow][jc];
#pragma unroll
          for (int gg = 0; gg < HEADS; ++gg) mix[gg] += pv * Wmix[hh][gg];
        }
        float mu = 0.f, s2 = 0.f;
#pragma unroll
        for (int gg = 0; gg < HEADS; ++gg) { mu += mix[gg]; s2 += mix[gg] * mix[gg]; }
        mu *= (1.f / HEADS);
        s2 = s2 * (1.f / HEADS) - mu * mu;
        float a = (mix[h] - mu) * rsqrtf(s2 + EPSV) * sg[h] + sb[h];
        ap[ms][e] = (bf16)a;
      }
    }
    __syncthreads();
    v16bf bv[4];
#pragma unroll
    for (int ns = 0; ns < 4; ++ns) {
      const bf16* base = Vh + (size_t)(ns * 16 + np) * M_TOT + j0;
      bv[ns] = load_frag(base + s * 8, base + 16 + s * 8);
    }
#pragma unroll
    for (int ms = 0; ms < 2; ++ms)
#pragma unroll
      for (int ns = 0; ns < 4; ++ns)
        acc[ms][ns] = __builtin_amdgcn_wmma_f32_16x16x32_bf16(
            false, ap[ms], false, bv[ns], (short)0, acc[ms][ns], false, false);
  }
  // store: [row][h*64 + dc] bf16
#pragma unroll
  for (int ms = 0; ms < 2; ++ms)
#pragma unroll
    for (int ns = 0; ns < 4; ++ns) {
      int dc = ns * 16 + np;
#pragma unroll
      for (int r = 0; r < 8; ++r) {
        int row = m0 + ms * 16 + s * 8 + r;
        attn_out[(size_t)row * INNER + h * DHEAD + dc] = (bf16)acc[ms][ns][r];
      }
    }
}

// ------------------------ output projection GEMM + bias (f32 out, tile 32x64)
__global__ __launch_bounds__(32) void k_oproj(const bf16* __restrict__ a_in,
                                              const bf16* __restrict__ wT,
                                              const float* __restrict__ bias,
                                              float* __restrict__ out) {
  int lane = threadIdx.x & 31;
  int np = lane & 15, s = lane >> 4;
  int m0 = blockIdx.x * 32, o0 = blockIdx.y * 64;
  v8f acc[2][4] = {};
  for (int kc = 0; kc < INNER; kc += 32) {
    v16bf a[2], bf[4];
#pragma unroll
    for (int ms = 0; ms < 2; ++ms) {
      const bf16* base = a_in + (size_t)(m0 + ms * 16 + np) * INNER + kc;
      a[ms] = load_frag(base + s * 8, base + 16 + s * 8);
    }
#pragma unroll
    for (int ns = 0; ns < 4; ++ns) {
      const bf16* base = wT + (size_t)(o0 + ns * 16 + np) * INNER + kc;
      bf[ns] = load_frag(base + s * 8, base + 16 + s * 8);
    }
#pragma unroll
    for (int ms = 0; ms < 2; ++ms)
#pragma unroll
      for (int ns = 0; ns < 4; ++ns)
        acc[ms][ns] = __builtin_amdgcn_wmma_f32_16x16x32_bf16(
            false, a[ms], false, bf[ns], (short)0, acc[ms][ns], false, false);
  }
#pragma unroll
  for (int ms = 0; ms < 2; ++ms)
#pragma unroll
    for (int ns = 0; ns < 4; ++ns) {
      int col = o0 + ns * 16 + np;
      float bb = bias[col];
#pragma unroll
      for (int r = 0; r < 8; ++r) {
        int row = m0 + ms * 16 + s * 8 + r;
        out[(size_t)row * DIM + col] = acc[ms][ns][r] + bb;
      }
    }
}

extern "C" void kernel_launch(void* const* d_in, const int* in_sizes, int n_in,
                              void* d_out, int out_size, void* d_ws, size_t ws_size,
                              hipStream_t stream) {
  (void)in_sizes; (void)n_in; (void)out_size; (void)ws_size;
  const float* x        = (const float*)d_in[0];
  const float* ln_g     = (const float*)d_in[1];
  const float* ln_b     = (const float*)d_in[2];
  const float* w_qkv    = (const float*)d_in[3];
  const float* reattn_w = (const float*)d_in[4];
  const float* rn_g     = (const float*)d_in[5];
  const float* rn_b     = (const float*)d_in[6];
  const float* w_out    = (const float*)d_in[7];
  const float* b_out    = (const float*)d_in[8];

  char* ws = (char*)d_ws;
  size_t off = 0;
  auto alloc = [&](size_t bytes) {
    void* p = ws + off;
    off += (bytes + 255) & ~(size_t)255;
    return p;
  };
  bf16* xn    = (bf16*)alloc((size_t)M_TOT * DIM * 2);
  bf16* wqkvT = (bf16*)alloc((size_t)NQKV * DIM * 2);
  bf16* woutT = (bf16*)alloc((size_t)DIM * INNER * 2);
  bf16* qb    = (bf16*)alloc((size_t)HEADS * M_TOT * DHEAD * 2);
  bf16* kb    = (bf16*)alloc((size_t)HEADS * M_TOT * DHEAD * 2);
  bf16* vTb   = (bf16*)alloc((size_t)HEADS * M_TOT * DHEAD * 2);
  bf16* aout  = (bf16*)alloc((size_t)M_TOT * INNER * 2);

  k_ln<<<M_TOT / 8, 256, 0, stream>>>(x, ln_g, ln_b, xn);
  k_transpose_bf16<<<512, 256, 0, stream>>>(w_qkv, wqkvT, DIM, NQKV);
  k_transpose_bf16<<<256, 256, 0, stream>>>(w_out, woutT, INNER, DIM);
  k_qkv<<<dim3(M_TOT / 32, NQKV / 64), 32, 0, stream>>>(xn, wqkvT, qb, kb, vTb);
  k_attn<<<M_TOT / 32, 256, 0, stream>>>(qb, kb, vTb, reattn_w, rn_g, rn_b, aout);
  k_oproj<<<dim3(M_TOT / 32, DIM / 64), 32, 0, stream>>>(aout, woutT, b_out, (float*)d_out);
}
